// GraphVertexRegistration_80066780332536
// MI455X (gfx1250) — matrix operand order repaired and checked by reading, not compiled, not run on hardware
//
#include <hip/hip_runtime.h>
#include <hip/hip_bf16.h>
#include <stdint.h>

typedef __attribute__((ext_vector_type(16))) _Float16 v16h;
typedef __attribute__((ext_vector_type(8)))  float    v8f;

#define TOPKN 16
#define EE 128

// ---------------------------------------------------------------------------
// Weight prep: fp32 -> fp16 (w_in zero-padded K 8->32 so one WMMA covers it)
// ---------------------------------------------------------------------------
__global__ __launch_bounds__(256)
void prep_kernel(const float* __restrict__ w_in,   // 128x8
                 const float* __restrict__ w_qkv,  // 384x128
                 const float* __restrict__ w_o,    // 128x128
                 _Float16* __restrict__ win_pad,   // 128x32
                 _Float16* __restrict__ wqkv_h,    // 384x128
                 _Float16* __restrict__ wo_h)      // 128x128
{
    int t = blockIdx.x * blockDim.x + threadIdx.x;
    if (t < 128 * 32) {
        int n = t >> 5, k = t & 31;
        win_pad[t] = (k < 8) ? (_Float16)w_in[n * 8 + k] : (_Float16)0.0f;
    }
    if (t < 384 * 128) wqkv_h[t] = (_Float16)w_qkv[t];
    if (t < 128 * 128) wo_h[t]   = (_Float16)w_o[t];
}

// ---------------------------------------------------------------------------
// Top-16 nearest neighbors by xy distance. One block per query box.
// 16 rounds of block-wide argmin; each thread owns j = tid + 256*s and
// excludes already-chosen via a private bitmask (N<=16384 -> s<64).
// ---------------------------------------------------------------------------
__global__ __launch_bounds__(256)
void topk_kernel(const float* __restrict__ boxes, int* __restrict__ topk, int N)
{
    __shared__ float svals[256];
    __shared__ int   sidx[256];
    const int i   = blockIdx.x;
    const int tid = threadIdx.x;
    const float cx = boxes[(size_t)i * 9 + 0];
    const float cy = boxes[(size_t)i * 9 + 1];
    uint64_t used = 0;

    for (int r = 0; r < TOPKN; ++r) {
        float best = 3.4e38f; int bj = -1;
        for (int s = 0;; ++s) {
            int j = tid + (s << 8);
            if (j >= N) break;
            if (used & (1ull << s)) continue;
            float dx = boxes[(size_t)j * 9 + 0] - cx;
            float dy = boxes[(size_t)j * 9 + 1] - cy;
            float d = dx * dx + dy * dy;
            if (d < best) { best = d; bj = j; }
        }
        svals[tid] = best; sidx[tid] = bj;
        __syncthreads();
        for (int off = 128; off > 0; off >>= 1) {
            if (tid < off && svals[tid + off] < svals[tid]) {
                svals[tid] = svals[tid + off];
                sidx[tid]  = sidx[tid + off];
            }
            __syncthreads();
        }
        int win = sidx[0];
        if (tid == 0) topk[(size_t)i * TOPKN + r] = win;
        if (win >= 0 && (win & 255) == tid) used |= 1ull << (win >> 8);
        __syncthreads();
    }
}

// ---------------------------------------------------------------------------
// WMMA fragment helpers (CDNA5 wave32 16-bit layouts, CDNA5 ISA)
//   A 16x32: lane -> M=lane&15, group g=lane>>4; elem j: K = g*16 + j (this file's order, not the ISA table)
//   B 32x16: lane -> N=lane&15, kg=lane>>4;      elem j: K = kg*16 + j (same order as A, not ISA table)
//   C 16x16 f32: lane -> N=lane&15, vgpr r -> M = r + (lane>=16 ? 8 : 0)
// ---------------------------------------------------------------------------
__device__ __forceinline__ v16h frag_zero16() {
    v16h a;
#pragma unroll
    for (int j = 0; j < 16; ++j) a[j] = (_Float16)0.0f;
    return a;
}

__device__ __forceinline__ v16h load_A32(const _Float16* base, int ld, int kbase) {
    const int lane = threadIdx.x & 31;
    const int m = lane & 15, g = lane >> 4;
    const _Float16* row = base + m * ld + kbase;
    v16h a;
#pragma unroll
    for (int j = 0; j < 16; ++j) a[j] = row[g * 16 + j];       // K = g*16 + j
    return a;
}

// A fragment for a 16x16 matrix zero-padded in K to 32 (K>=16 is zero)
__device__ __forceinline__ v16h load_A16(const _Float16* base, int ld, int col0) {
    const int lane = threadIdx.x & 31;
    const int m = lane & 15, g = lane >> 4;
    v16h a = frag_zero16();
    const _Float16* row = base + m * ld + col0;
    if (g == 0) {
#pragma unroll
        for (int j = 0; j < 16; ++j) a[j] = row[j];   // K = j  (0..15), lanes 16-31 zero
    }
    return a;
}

// B[k][n] = w[(nbase+n)*ld + kbase + k]  (i.e. B = W^T tile, W row-major)
__device__ __forceinline__ v16h load_Bt(const _Float16* w, int ld, int nbase, int kbase) {
    const int lane = threadIdx.x & 31;
    const int n = lane & 15, kg = lane >> 4;
    const _Float16* p = w + (size_t)(nbase + n) * ld + kbase + kg * 16;
    v16h b;
#pragma unroll
    for (int j = 0; j < 16; ++j) b[j] = p[j];
    return b;
}

__device__ __forceinline__ v8f acc_zero() {
    v8f c;
#pragma unroll
    for (int r = 0; r < 8; ++r) c[r] = 0.0f;
    return c;
}

__device__ __forceinline__ void store_C_f16(v8f c, _Float16* dst, int ld, int n0,
                                            const float* bias, bool relu) {
    const int lane = threadIdx.x & 31;
    const int n = lane & 15, mb = (lane >> 4) * 8;
    const float b = bias ? bias[n0 + n] : 0.0f;
#pragma unroll
    for (int r = 0; r < 8; ++r) {
        float v = c[r] + b;
        if (relu) v = fmaxf(v, 0.0f);
        dst[(mb + r) * ld + n0 + n] = (_Float16)v;
    }
}

#define WMMA_F16(a, b, c) \
    __builtin_amdgcn_wmma_f32_16x16x32_f16(false, (a), false, (b), (short)0, (c), false, false)

// ---------------------------------------------------------------------------
// Main per-box kernel: feat -> MLP -> 8-head attention -> proj -> pool -> out
// One block (128 threads = 4 waves) per box. Everything staged in LDS.
// ---------------------------------------------------------------------------
__global__ __launch_bounds__(128)
void nbr_attn_kernel(const float* __restrict__ boxes,
                     const int*  __restrict__ topk,
                     const _Float16* __restrict__ win_pad,  // 128x32
                     const float* __restrict__ b_in,
                     const _Float16* __restrict__ wqkv_h,   // 384x128
                     const float* __restrict__ b_qkv,
                     const _Float16* __restrict__ wo_h,     // 128x128
                     const float* __restrict__ b_o,
                     const float* __restrict__ w_out,       // 128x256 (f32)
                     const float* __restrict__ b_out,
                     float* __restrict__ out, int N)
{
    __shared__ _Float16 sFeat[16 * 32];     // feat, K-padded to 32
    __shared__ _Float16 sX[16 * 128];       // relu(feat @ Win^T)
    __shared__ _Float16 sQKV[16 * 384];     // q|k|v
    __shared__ _Float16 sAtt[16 * 128];     // attention output (pre-Wo)
    __shared__ float    sOut[16 * 128];     // after Wo (f32, for pooling)
    __shared__ float    sPool[256];         // [mean(128) | max(128)]
    __shared__ _Float16 sProb[4][16 * 16];  // per-wave softmax scratch

    const int i    = blockIdx.x;
    const int tid  = threadIdx.x;
    const int wid  = tid >> 5;
    const int lane = tid & 31;

    // ---- Phase A: geometric features -> sFeat (16 x 32, zero padded) ----
#pragma unroll
    for (int r = 0; r < 4; ++r) sFeat[tid * 4 + r] = (_Float16)0.0f;
    __syncthreads();

    if (tid < 16) {
        const int j = topk[(size_t)i * TOPKN + tid];
        const float* cb = boxes + (size_t)i * 9;
        const float* nb = boxes + (size_t)j * 9;
        const float cx = cb[0], cy = cb[1], ch = cb[6];
        const float dx = nb[0] - cx, dy = nb[1] - cy;
        const float ea = atan2f(dy, dx) - ch;
        const float ed = sqrtf(dx * dx + dy * dy) * 0.01f;
        const float na = nb[6] - ch;
        _Float16* f = sFeat + tid * 32;
        f[0] = (_Float16)ed;
        f[1] = (_Float16)sinf(ea);
        f[2] = (_Float16)cosf(ea);
        f[3] = (_Float16)sinf(na);
        f[4] = (_Float16)cosf(na);
        f[5] = (_Float16)(nb[3] * 0.2f);
        f[6] = (_Float16)(nb[4] * 0.2f);
        f[7] = (_Float16)(nb[5] * 0.2f);
    }
    __syncthreads();

    // ---- Phase B: x = relu(feat @ Win^T + b_in)   (16x32)·(32x128) ----
    {
        v16h a = load_A32(sFeat, 32, 0);
#pragma unroll
        for (int t = 0; t < 2; ++t) {
            const int n0 = (wid + t * 4) * 16;
            v16h b = load_Bt(win_pad, 32, n0, 0);
            v8f  c = acc_zero();
            c = WMMA_F16(a, b, c);
            store_C_f16(c, sX, 128, n0, b_in, true);
        }
    }
    __syncthreads();

    // ---- Phase C: qkv = x @ Wqkv^T + b_qkv   (16x128)·(128x384) ----
    for (int t = wid; t < 24; t += 4) {
        const int n0 = t * 16;
        v8f c = acc_zero();
#pragma unroll
        for (int kk = 0; kk < 4; ++kk) {
            v16h a = load_A32(sX, 128, kk * 32);
            v16h b = load_Bt(wqkv_h, 128, n0, kk * 32);
            c = WMMA_F16(a, b, c);
        }
        store_C_f16(c, sQKV, 384, n0, b_qkv, false);
    }
    __syncthreads();

    // ---- Phase D: 8-head attention (head dim 16, 16 keys) ----
#pragma unroll
    for (int hh = 0; hh < 2; ++hh) {
        const int h = wid + hh * 4;
        // scores = q_h @ k_h^T * 0.25
        v16h aq = load_A16(sQKV, 384, h * 16);         // q rows, K padded
        v16h bk = frag_zero16();                        // B = k^T (K<16 valid)
        if ((lane >> 4) == 0) {
            const int n = lane & 15;
            const _Float16* p = sQKV + n * 384 + 128 + h * 16;
#pragma unroll
            for (int j = 0; j < 16; ++j) bk[j] = p[j];
        }
        v8f c = acc_zero();
        c = WMMA_F16(aq, bk, c);

        // softmax across k (the N dim of C = across lanes in 16-lane halves)
#pragma unroll
        for (int r = 0; r < 8; ++r) {
            float v = c[r] * 0.25f;
            float mx = v;
            mx = fmaxf(mx, __shfl_xor(mx, 1));
            mx = fmaxf(mx, __shfl_xor(mx, 2));
            mx = fmaxf(mx, __shfl_xor(mx, 4));
            mx = fmaxf(mx, __shfl_xor(mx, 8));
            float e = __expf(v - mx);
            float s = e;
            s += __shfl_xor(s, 1);
            s += __shfl_xor(s, 2);
            s += __shfl_xor(s, 4);
            s += __shfl_xor(s, 8);
            c[r] = e / s;
        }
        // stash probs (transpose lanes->rows via LDS, per-wave scratch)
        {
            const int n = lane & 15, mb = (lane >> 4) * 8;
#pragma unroll
            for (int r = 0; r < 8; ++r)
                sProb[wid][(mb + r) * 16 + n] = (_Float16)c[r];
        }
        // out_h = probs @ v_h   (16x16)·(16x16), K padded to 32
        v16h ap = load_A16(sProb[wid], 16, 0);
        v16h bv = frag_zero16();
        if ((lane >> 4) == 0) {
            const int n = lane & 15;
#pragma unroll
            for (int j = 0; j < 16; ++j)
                bv[j] = sQKV[j * 384 + 256 + h * 16 + n];
        }
        v8f co = acc_zero();
        co = WMMA_F16(ap, bv, co);
        store_C_f16(co, sAtt, 128, h * 16, nullptr, false);
    }
    __syncthreads();

    // ---- Phase E: out2 = att @ Wo^T + b_o   (16x128)·(128x128), f32 out ----
#pragma unroll
    for (int t = 0; t < 2; ++t) {
        const int n0 = (wid + t * 4) * 16;
        v8f c = acc_zero();
#pragma unroll
        for (int kk = 0; kk < 4; ++kk) {
            v16h a = load_A32(sAtt, 128, kk * 32);
            v16h b = load_Bt(wo_h, 128, n0, kk * 32);
            c = WMMA_F16(a, b, c);
        }
        const int n = lane & 15, mb = (lane >> 4) * 8;
        const float bb = b_o[n0 + n];
#pragma unroll
        for (int r = 0; r < 8; ++r)
            sOut[(mb + r) * 128 + n0 + n] = c[r] + bb;
    }
    __syncthreads();

    // ---- Phase F: mean/max pool over the 16 neighbors ----
    {
        float s = 0.0f, mx = -3.4e38f;
#pragma unroll
        for (int r = 0; r < 16; ++r) {
            float v = sOut[r * 128 + tid];
            s += v;
            mx = fmaxf(mx, v);
        }
        sPool[tid]       = s * (1.0f / 16.0f);
        sPool[128 + tid] = mx;
    }
    __syncthreads();

    // ---- Phase G: y = pooled @ Wout^T + b_out  (1x256 matvec, VALU) ----
    {
        float acc = b_out[tid];
        const float* wr = w_out + (size_t)tid * 256;
#pragma unroll 8
        for (int j = 0; j < 256; ++j) acc += sPool[j] * wr[j];
        out[(size_t)i * 128 + tid] = acc;
    }
}

// ---------------------------------------------------------------------------
extern "C" void kernel_launch(void* const* d_in, const int* in_sizes, int n_in,
                              void* d_out, int out_size, void* d_ws, size_t ws_size,
                              hipStream_t stream)
{
    const float* boxes = (const float*)d_in[0];
    const float* w_in  = (const float*)d_in[1];
    const float* b_in  = (const float*)d_in[2];
    const float* w_qkv = (const float*)d_in[3];
    const float* b_qkv = (const float*)d_in[4];
    const float* w_o   = (const float*)d_in[5];
    const float* b_o   = (const float*)d_in[6];
    const float* w_out = (const float*)d_in[7];
    const float* b_out = (const float*)d_in[8];

    const int N = in_sizes[0] / 9;

    char* ws = (char*)d_ws;
    size_t off = 0;
    int* topk = (int*)(ws + off);
    off += (size_t)N * TOPKN * sizeof(int);
    off = (off + 255) & ~(size_t)255;
    _Float16* win_pad = (_Float16*)(ws + off); off += 128 * 32  * sizeof(_Float16);
    _Float16* wqkv_h  = (_Float16*)(ws + off); off += 384 * 128 * sizeof(_Float16);
    _Float16* wo_h    = (_Float16*)(ws + off); off += 128 * 128 * sizeof(_Float16);

    prep_kernel<<<(384 * 128 + 255) / 256, 256, 0, stream>>>(
        w_in, w_qkv, w_o, win_pad, wqkv_h, wo_h);

    topk_kernel<<<N, 256, 0, stream>>>(boxes, topk, N);

    nbr_attn_kernel<<<N, 128, 0, stream>>>(
        boxes, topk, win_pad, b_in, wqkv_h, b_qkv, wo_h, b_o,
        w_out, b_out, (float*)d_out, N);
}
